// SelfAttention_64871186039054
// MI455X (gfx1250) — matrix operand order repaired and checked by reading, not compiled, not run on hardware
//
#include <hip/hip_runtime.h>

// ---------------------------------------------------------------------------
// MI455X (gfx1250) self-attention layer, bf16 WMMA pipeline with async
// global->LDS double-buffered staging.
// ---------------------------------------------------------------------------

typedef unsigned short u16;
typedef unsigned int   u32;
typedef __attribute__((ext_vector_type(16))) __bf16 bf16x16;
typedef __attribute__((ext_vector_type(8)))  float  f32x8;
typedef __attribute__((ext_vector_type(4)))  int    i32x4;

// address-space-qualified vector types for the async global->LDS builtin
typedef __attribute__((address_space(1))) i32x4 as1_i32x4;
typedef __attribute__((address_space(3))) i32x4 as3_i32x4;

#if __has_builtin(__builtin_amdgcn_global_load_async_to_lds_b128)
#define HAVE_ASYNC_LDS 1
#endif

union FragB { bf16x16 v; u32 u[8]; };

static __device__ __forceinline__ u16 f2bf(float f) {
    u32 x = __float_as_uint(f);
    return (u16)((x + 0x7FFFu + ((x >> 16) & 1u)) >> 16);   // round-to-nearest-even
}
static __device__ __forceinline__ float bf2f(u16 h) {
    return __uint_as_float(((u32)h) << 16);
}
static __device__ __forceinline__ f32x8 wmma_bf16(const FragB& a, const FragB& b, f32x8 c) {
    return __builtin_amdgcn_wmma_f32_16x16x32_bf16(false, a.v, false, b.v,
                                                   (short)0, c, false, false);
}

// 16-byte global -> LDS copy; async (ASYNCcnt) when the toolchain exposes it.
static __device__ __forceinline__ void cp16(const u16* g, u16* l) {
#ifdef HAVE_ASYNC_LDS
    __builtin_amdgcn_global_load_async_to_lds_b128(
        (as1_i32x4*)(u16*)g, (as3_i32x4*)l, 0, 0);
#else
    uint4 d = *(const uint4*)g;
    *(uint4*)l = d;
#endif
}
static __device__ __forceinline__ void wait_async() {
#ifdef HAVE_ASYNC_LDS
    asm volatile("s_wait_asynccnt 0x0" ::: "memory");
#endif
}

// ---------------------------------------------------------------------------
// Kernel 0a: f32 -> bf16 conversion (vectorized x4)
// ---------------------------------------------------------------------------
__global__ void cvt_f32_bf16(const float* __restrict__ in, u16* __restrict__ out, int n4) {
    int i = blockIdx.x * 256 + threadIdx.x;
    if (i >= n4) return;
    float4 f = ((const float4*)in)[i];
    u32 lo = (u32)f2bf(f.x) | ((u32)f2bf(f.y) << 16);
    u32 hi = (u32)f2bf(f.z) | ((u32)f2bf(f.w) << 16);
    ((uint2*)out)[i] = make_uint2(lo, hi);
}

// ---------------------------------------------------------------------------
// Kernel 0b: f32 [K][N] -> bf16 [N][K] transpose (32x32 LDS tiles).
// Lets the GEMM stage its B operand as straight row-major async copies.
// ---------------------------------------------------------------------------
__global__ __launch_bounds__(256) void cvt_transpose(const float* __restrict__ in,
                                                     u16* __restrict__ out, int K, int N) {
    __shared__ float tile[32][33];
    const int tx = threadIdx.x & 31, tg = threadIdx.x >> 5;   // 8 groups of 4 rows
    const int k0 = blockIdx.y * 32, n0 = blockIdx.x * 32;
#pragma unroll
    for (int j = 0; j < 4; ++j) {
        int r = tg * 4 + j;
        tile[r][tx] = in[(size_t)(k0 + r) * N + n0 + tx];
    }
    __syncthreads();
#pragma unroll
    for (int j = 0; j < 4; ++j) {
        int r = tg * 4 + j;
        out[(size_t)(n0 + r) * K + k0 + tx] = f2bf(tile[tx][r]);
    }
}

// ---------------------------------------------------------------------------
// Kernel 1/4: 128x128-tile bf16 GEMM, A[m][k] x Bt[n][k], double-buffered
// async LDS staging, 8 waves, 16x16x32 WMMA.
// mode 0: scatter into q (x0.125) / k (B,H,N,HD) and v (B,H,HD,N) bf16.
// mode 1: add bias, write f32.
// ---------------------------------------------------------------------------
__global__ __launch_bounds__(256, 1) void gemm128(
    const u16* __restrict__ A, const u16* __restrict__ Bt, int K, int mode,
    u16* __restrict__ qo, u16* __restrict__ ko, u16* __restrict__ vo,
    const float* __restrict__ bias, float* __restrict__ outf)
{
    __shared__ u16 ldsA[2][128 * 32];   // [m][k] row-major
    __shared__ u16 ldsB[2][128 * 32];   // [n][k] row-major

    const int t = threadIdx.x;
    const int lane = t & 31, wv = t >> 5;
    const int half = lane >> 4, l15 = lane & 15;
    const int wm = wv & 3, wn = wv >> 2;
    const int m0 = blockIdx.y * 128, n0 = blockIdx.x * 128;

    const f32x8 z = {0.f, 0.f, 0.f, 0.f, 0.f, 0.f, 0.f, 0.f};
    f32x8 acc[2][4];
#pragma unroll
    for (int mt = 0; mt < 2; ++mt)
#pragma unroll
        for (int nt = 0; nt < 4; ++nt) acc[mt][nt] = z;

    const int sr = t >> 1, sc = (t & 1) * 8;   // staging: row, dword-col (8 dwords = 2 chunks)
    const u16* gA = A + (size_t)(m0 + sr) * K + sc * 2;
    const u16* gB = Bt + (size_t)(n0 + sr) * K + sc * 2;

    auto stage = [&](int kt, int buf) {
#pragma unroll
        for (int c = 0; c < 2; ++c) {
            cp16(gA + kt * 32 + c * 8, &ldsA[buf][sr * 32 + sc * 2 + c * 8]);
            cp16(gB + kt * 32 + c * 8, &ldsB[buf][sr * 32 + sc * 2 + c * 8]);
        }
#ifndef HAVE_ASYNC_LDS
        if ((kt + 1) * 32 < K) {
            __builtin_prefetch(gA + (kt + 1) * 32, 0, 3);
            __builtin_prefetch(gB + (kt + 1) * 32, 0, 3);
        }
#endif
    };

    const int T = K >> 5;
    stage(0, 0);
    wait_async();
    __syncthreads();

    for (int kt = 0; kt < T; ++kt) {
        const int buf = kt & 1;
        if (kt + 1 < T) stage(kt + 1, buf ^ 1);

        const u32* adw = (const u32*)ldsA[buf];
        const u32* bdw = (const u32*)ldsB[buf];
        FragB af[2], bf[4];
#pragma unroll
        for (int mt = 0; mt < 2; ++mt) {
            const int row = wm * 32 + mt * 16 + l15;
#pragma unroll
            for (int i = 0; i < 8; ++i)   // A layout: K=(i>>2)*16+half*8+(i&3)*2
                af[mt].u[i] = adw[row * 16 + (i >> 2) * 8 + half * 4 + (i & 3)];
        }
#pragma unroll
        for (int nt = 0; nt < 4; ++nt) {
            const int n = wn * 64 + nt * 16 + l15;
#pragma unroll
            for (int i = 0; i < 8; ++i)   // B layout: K=(i>>2)*16+half*8+(i&3)*2
                bf[nt].u[i] = bdw[n * 16 + (i >> 2) * 8 + half * 4 + (i & 3)];
        }
#pragma unroll
        for (int mt = 0; mt < 2; ++mt)
#pragma unroll
            for (int nt = 0; nt < 4; ++nt)
                acc[mt][nt] = wmma_bf16(af[mt], bf[nt], acc[mt][nt]);

        wait_async();
        __syncthreads();
    }

    // epilogue
#pragma unroll
    for (int mt = 0; mt < 2; ++mt)
#pragma unroll
        for (int nt = 0; nt < 4; ++nt) {
            const int c = n0 + wn * 64 + nt * 16 + l15;
#pragma unroll
            for (int r = 0; r < 8; ++r) {
                const int row = m0 + wm * 32 + mt * 16 + r + half * 8;  // C/D: M=r+8*half
                float val = acc[mt][nt][r];
                if (mode == 0) {
                    const int which = c >> 10, h = (c >> 6) & 15, hd = c & 63;
                    const int b = row >> 11, p = row & 2047;
                    const size_t bh = (size_t)(b * 16 + h);
                    if (which == 0) {        // q, scaled by 1/sqrt(HD) (exact in bf16)
                        qo[(bh * 2048 + p) * 64 + hd] = f2bf(val * 0.125f);
                    } else if (which == 1) { // k, [b,h,n,hd]
                        ko[(bh * 2048 + p) * 64 + hd] = f2bf(val);
                    } else {                 // v stored transposed: [b,h,hd,n]
                        vo[(bh * 64 + hd) * 2048 + p] = f2bf(val);
                    }
                } else {
                    outf[(size_t)row * 1024 + c] = val + bias[c];
                }
            }
        }
}

// ---------------------------------------------------------------------------
// Kernel 2: RoPE on the suffix positions of q and k (bf16 in place).
// ---------------------------------------------------------------------------
__global__ void rope_k(u16* __restrict__ q, u16* __restrict__ k,
                       const float* __restrict__ sinp, const float* __restrict__ cosp,
                       int prefix, int n_rope, int total)
{
    int tid = blockIdx.x * 256 + threadIdx.x;
    if (tid >= total) return;
    int i = tid & 31;                 // pair index within head dim (0..31)
    int rest = tid >> 5;
    int r = rest % n_rope;            // rope table row
    int bh = rest / n_rope;           // (b*H + h)
    size_t base = ((size_t)bh * 2048 + (prefix + r)) * 64;
    float c1 = cosp[r * 64 + i],      s1 = sinp[r * 64 + i];
    float c2 = cosp[r * 64 + i + 32], s2 = sinp[r * 64 + i + 32];
    float aq = bf2f(q[base + i]), bq = bf2f(q[base + i + 32]);
    q[base + i]      = f2bf(aq * c1 - bq * s1);
    q[base + i + 32] = f2bf(bq * c2 + aq * s2);
    float ak = bf2f(k[base + i]), bk = bf2f(k[base + i + 32]);
    k[base + i]      = f2bf(ak * c1 - bk * s1);
    k[base + i + 32] = f2bf(bk * c2 + ak * s2);
}

// ---------------------------------------------------------------------------
// Kernel 3: flash attention. One block = 256 query rows of one (b,h);
// each wave owns 32 query rows; 64-key K/V tiles double-buffered in LDS via
// async copies (V arrives pre-transposed [hd][key] from the QKV epilogue).
// ---------------------------------------------------------------------------
__global__ __launch_bounds__(256, 1) void attn64(
    const u16* __restrict__ q, const u16* __restrict__ k, const u16* __restrict__ v,
    u16* __restrict__ ob)
{
    __shared__ u16 ldsK[2][64 * 64];     // [key][hd]
    __shared__ u16 ldsV[2][64 * 64];     // [hd][key]
    __shared__ u16 ldsP[8][2][16 * 32];  // per-wave P relayout (C-format -> A-format)

    const int t = threadIdx.x, lane = t & 31, wv = t >> 5;
    const int half = lane >> 4, l15 = lane & 15;
    const int bh = blockIdx.y, b = bh >> 4, h = bh & 15;
    const size_t hbase = (size_t)bh * 2048 * 64;
    const int q0 = blockIdx.x * 256 + wv * 32;

    // Q fragments (A-operand), 32 rows x 64 hd, loaded straight from global.
    FragB qf[2][2];
#pragma unroll
    for (int mt = 0; mt < 2; ++mt) {
        const u32* qr = (const u32*)(q + hbase + (size_t)(q0 + mt * 16 + l15) * 64);
#pragma unroll
        for (int s = 0; s < 2; ++s)
#pragma unroll
            for (int i = 0; i < 8; ++i)
                qf[mt][s].u[i] = qr[s * 16 + (i >> 2) * 8 + half * 4 + (i & 3)];
    }

    const f32x8 z = {0.f, 0.f, 0.f, 0.f, 0.f, 0.f, 0.f, 0.f};
    f32x8 oacc[2][4];
    float mrow[2][8], lrow[2][8];
#pragma unroll
    for (int mt = 0; mt < 2; ++mt) {
#pragma unroll
        for (int nt = 0; nt < 4; ++nt) oacc[mt][nt] = z;
#pragma unroll
        for (int r = 0; r < 8; ++r) { mrow[mt][r] = -1e30f; lrow[mt][r] = 0.f; }
    }

    // staging assignment: 8 dwords (2 chunks) per thread per matrix
    const int sd = t * 8;
    const int skey = sd >> 5, sdc = sd & 31;   // K: key row, dword within 64-hd row
    const u16* gK = k + hbase + (size_t)skey * 64 + sdc * 2;     // row stride 64 halfs
    const u16* gV = v + hbase + (size_t)skey * 2048 + sdc * 2;   // [hd][key], row stride 2048

    auto stage = [&](int j0, int buf) {
#pragma unroll
        for (int c = 0; c < 2; ++c) {
            cp16(gK + (size_t)j0 * 64 + c * 8, &ldsK[buf][sd * 2 + c * 8]);   // key-major tile
            cp16(gV + j0 + c * 8,              &ldsV[buf][sd * 2 + c * 8]);   // hd-major tile
        }
    };

    stage(0, 0);
    wait_async();
    __syncthreads();

    for (int jt = 0; jt < 32; ++jt) {
        const int buf = jt & 1;
        if (jt + 1 < 32) stage((jt + 1) * 64, buf ^ 1);

        const u32* ldsKdw = (const u32*)ldsK[buf];
        const u32* ldsVdw = (const u32*)ldsV[buf];

#pragma unroll
        for (int c = 0; c < 2; ++c) {           // two 32-key chunks
            // S = Q · K^T  (32 rows x 32 keys per wave)
            f32x8 sc[2][2];
#pragma unroll
            for (int mt = 0; mt < 2; ++mt)
#pragma unroll
                for (int j = 0; j < 2; ++j) sc[mt][j] = z;
#pragma unroll
            for (int j = 0; j < 2; ++j)
#pragma unroll
                for (int s = 0; s < 2; ++s) {
                    FragB kb;
#pragma unroll
                    for (int i = 0; i < 8; ++i)
                        kb.u[i] = ldsKdw[(c * 32 + j * 16 + l15) * 32 + s * 16 + (i >> 2) * 8 + half * 4 + (i & 3)];
#pragma unroll
                    for (int mt = 0; mt < 2; ++mt)
                        sc[mt][j] = wmma_bf16(qf[mt][s], kb, sc[mt][j]);
                }

            // online softmax per row (rows live in 16-lane half-groups)
#pragma unroll
            for (int mt = 0; mt < 2; ++mt) {
                float alpha[8];
#pragma unroll
                for (int r = 0; r < 8; ++r) {
                    float s0 = sc[mt][0][r], s1 = sc[mt][1][r];
                    float mx = fmaxf(s0, s1);
#pragma unroll
                    for (int msk = 1; msk <= 8; msk <<= 1)
                        mx = fmaxf(mx, __shfl_xor(mx, msk, 32));
                    float mnew = fmaxf(mrow[mt][r], mx);
                    float al = __expf(mrow[mt][r] - mnew);
                    float p0 = __expf(s0 - mnew), p1 = __expf(s1 - mnew);
                    float rs = p0 + p1;
#pragma unroll
                    for (int msk = 1; msk <= 8; msk <<= 1)
                        rs += __shfl_xor(rs, msk, 32);
                    lrow[mt][r] = lrow[mt][r] * al + rs;
                    mrow[mt][r] = mnew;
                    alpha[r] = al;
                    const int m = r + half * 8;               // C-format row
                    ldsP[wv][mt][m * 32 + l15]      = f2bf(p0);
                    ldsP[wv][mt][m * 32 + 16 + l15] = f2bf(p1);
                }
#pragma unroll
                for (int nt = 0; nt < 4; ++nt)
#pragma unroll
                    for (int r = 0; r < 8; ++r) oacc[mt][nt][r] *= alpha[r];
            }

            // in-wave LDS store->load ordering (split dep counter)
            asm volatile("s_wait_dscnt 0x0" ::: "memory");

            // O += P · V
            FragB pf[2];
#pragma unroll
            for (int mt = 0; mt < 2; ++mt) {
                const u32* pd = (const u32*)(&ldsP[wv][mt][0]);
#pragma unroll
                for (int i = 0; i < 8; ++i)
                    pf[mt].u[i] = pd[l15 * 16 + (i >> 2) * 8 + half * 4 + (i & 3)];
            }
#pragma unroll
            for (int nt = 0; nt < 4; ++nt) {
                FragB vb;
#pragma unroll
                for (int i = 0; i < 8; ++i)
                    vb.u[i] = ldsVdw[(nt * 16 + l15) * 32 + c * 16 + (i >> 2) * 8 + half * 4 + (i & 3)];
#pragma unroll
                for (int mt = 0; mt < 2; ++mt)
                    oacc[mt][nt] = wmma_bf16(pf[mt], vb, oacc[mt][nt]);
            }
        }

        wait_async();
        __syncthreads();
    }

    // normalize and write O as bf16 in (B, N, D) layout for the projection GEMM
#pragma unroll
    for (int mt = 0; mt < 2; ++mt) {
        float inv[8];
#pragma unroll
        for (int r = 0; r < 8; ++r) inv[r] = 1.0f / lrow[mt][r];
#pragma unroll
        for (int nt = 0; nt < 4; ++nt) {
            const int col = h * 64 + nt * 16 + l15;
#pragma unroll
            for (int r = 0; r < 8; ++r) {
                const int row = q0 + mt * 16 + r + half * 8;
                ob[((size_t)b * 2048 + row) * 1024 + col] = f2bf(oacc[mt][nt][r] * inv[r]);
            }
        }
    }
}

// ---------------------------------------------------------------------------
// Host launcher
// ---------------------------------------------------------------------------
extern "C" void kernel_launch(void* const* d_in, const int* in_sizes, int n_in,
                              void* d_out, int out_size, void* d_ws, size_t ws_size,
                              hipStream_t stream) {
    (void)n_in; (void)out_size; (void)ws_size;
    const float* x     = (const float*)d_in[0];
    const float* sinp  = (const float*)d_in[1];
    const float* cosp  = (const float*)d_in[2];
    const float* Wqkv  = (const float*)d_in[3];
    const float* Wproj = (const float*)d_in[4];
    const float* bias  = (const float*)d_in[5];
    float* outp = (float*)d_out;

    const int Bb = 2, Nn = 2048, Dd = 1024, Hh = 16, HD = 64;
    const int n_rope = in_sizes[1] / HD;        // 2043
    const int prefix = Nn - n_rope;             // 5

    // workspace carving (256B aligned)
    char* w = (char*)d_ws;
    size_t off = 0;
    auto carve = [&](size_t bytes) -> char* {
        char* p = w + off;
        off = (off + bytes + 255) & ~(size_t)255;
        return p;
    };
    u16* xb   = (u16*)carve((size_t)Bb * Nn * Dd * 2);        // x bf16 [row][k]
    u16* wqb  = (u16*)carve((size_t)Dd * 3 * Dd * 2);         // W_qkv^T bf16 [n][k]
    u16* wpb  = (u16*)carve((size_t)Dd * Dd * 2);             // W_proj^T bf16 [n][k]
    u16* qb   = (u16*)carve((size_t)Bb * Hh * Nn * HD * 2);   // q (scaled) [b,h,n,hd]
    u16* kb   = (u16*)carve((size_t)Bb * Hh * Nn * HD * 2);   // k [b,h,n,hd]
    u16* vb   = (u16*)carve((size_t)Bb * Hh * Nn * HD * 2);   // v [b,h,hd,n] (transposed)
    u16* obuf = (u16*)carve((size_t)Bb * Nn * Dd * 2);        // attention output [row][d]

    // 0) convert to bf16 (+ transpose weights so GEMM B staging is row-major)
    {
        int n4 = Bb * Nn * Dd / 4;
        cvt_f32_bf16<<<(n4 + 255) / 256, 256, 0, stream>>>(x, xb, n4);
        cvt_transpose<<<dim3(3 * Dd / 32, Dd / 32), 256, 0, stream>>>(Wqkv, wqb, Dd, 3 * Dd);
        cvt_transpose<<<dim3(Dd / 32, Dd / 32), 256, 0, stream>>>(Wproj, wpb, Dd, Dd);
    }
    // 1) QKV GEMM: (4096 x 1024) x (1024 x 3072) -> scatter into q/k/v
    gemm128<<<dim3(3 * Dd / 128, Bb * Nn / 128), 256, 0, stream>>>(
        xb, wqb, Dd, 0, qb, kb, vb, nullptr, nullptr);
    // 2) RoPE on q/k suffix
    {
        int total = Bb * Hh * n_rope * 32;
        rope_k<<<(total + 255) / 256, 256, 0, stream>>>(qb, kb, sinp, cosp,
                                                        prefix, n_rope, total);
    }
    // 3) flash attention
    attn64<<<dim3(Nn / 256, Bb * Hh), 256, 0, stream>>>(qb, kb, vb, obuf);
    // 4) output projection + bias: (4096 x 1024) x (1024 x 1024) -> f32 out
    gemm128<<<dim3(Dd / 128, Bb * Nn / 128), 256, 0, stream>>>(
        obuf, wpb, Dd, 1, nullptr, nullptr, nullptr, bias, outp);
}
